// Transformer_55319178773070
// MI455X (gfx1250) — matrix operand order repaired and checked by reading, not compiled, not run on hardware
//
#include <hip/hip_runtime.h>
#include <hip/hip_bf16.h>

typedef __bf16 bf16;
typedef __attribute__((ext_vector_type(16))) __bf16 v16bf;
typedef __attribute__((ext_vector_type(8)))  __bf16 v8bf;
typedef __attribute__((ext_vector_type(8)))  float  v8f;

#define B_   4
#define T_   2048
#define C_   512
#define H_   8
#define HS   64
#define NTOK (B_ * T_)            // 8192 tokens
#define SM_SCALE 12.5f            // (1/sqrt(64)) / 0.01

static __device__ __forceinline__ v8f wmma_bf16(v16bf a, v16bf b, v8f c) {
  return __builtin_amdgcn_wmma_f32_16x16x32_bf16(false, a, false, b, (short)0, c,
                                                 false, false);
}

// A-fragment (16x32 bf16, MxK): lane m=lane%16, hf=lane/16.
// elements 0..7  <- row[k0 + hf*16 .. +7]
// elements 8..15 <- row[k0 + hf*16 + 8 .. +7]        (contiguous k order, matches B)
static __device__ __forceinline__ v16bf load_a_frag(const bf16* __restrict__ row,
                                                    int k0, int hf) {
  v8bf lo = *(const v8bf*)(row + k0 + hf * 16);
  v8bf hi = *(const v8bf*)(row + k0 + hf * 16 + 8);
  v16bf r;
#pragma unroll
  for (int i = 0; i < 8; ++i) { r[i] = lo[i]; r[i + 8] = hi[i]; }
  return r;
}

// B-fragment (32x16 bf16, KxN) from a row-major [N][K] operand:
// lane n=lane%16, hf=lane/16; element i -> k = k0 + hf*16 + i  (contiguous 32B)
static __device__ __forceinline__ v16bf load_b_frag(const bf16* __restrict__ row,
                                                    int k0, int hf) {
  return *(const v16bf*)(row + k0 + hf * 16);
}

// ---------------------------------------------------------------------------
// fp32 -> bf16 conversion of x, w_attn, w_proj; also writes att_l1 = +inf
// ---------------------------------------------------------------------------
__global__ void convert_kernel(const float* __restrict__ x,
                               const float* __restrict__ wa,
                               const float* __restrict__ wp,
                               bf16* __restrict__ xb, bf16* __restrict__ wab,
                               bf16* __restrict__ wpb, float* __restrict__ l1out) {
  int i = blockIdx.x * blockDim.x + threadIdx.x;
  int stride = gridDim.x * blockDim.x;
  for (int j = i; j < NTOK * C_; j += stride) xb[j] = (bf16)x[j];
  for (int j = i; j < 3 * C_ * C_; j += stride) wab[j] = (bf16)wa[j];
  for (int j = i; j < C_ * C_; j += stride) wpb[j] = (bf16)wp[j];
  if (i == 0) *l1out = __builtin_inff();  // mean|att| over masked (-inf) scores
}

// ---------------------------------------------------------------------------
// qkv = x @ w_attn^T ; scatter into q[B,H,T,hs], k[B,H,T,hs], vT[B,H,hs,T] bf16
// One wave computes a 32(M) x 64(N) tile; ping-pong K pipeline (no rotation
// movs). 256 * 24 = 6144 waves.
// ---------------------------------------------------------------------------
__global__ __launch_bounds__(256) void qkv_gemm_kernel(
    const bf16* __restrict__ xb, const bf16* __restrict__ wab,
    bf16* __restrict__ q, bf16* __restrict__ kk, bf16* __restrict__ vT) {
  const int lane = threadIdx.x & 31;
  const int wid = (blockIdx.x * blockDim.x + threadIdx.x) >> 5;
  const int NT = (3 * C_) / 64;  // 24 column chunks
  const int mt = wid / NT;       // 32-row tiles
  const int n0 = (wid % NT) * 64;
  if (mt >= NTOK / 32) return;
  const int l16 = lane & 15, hf = lane >> 4;
  const bf16* ar0 = xb + (size_t)(mt * 32 + l16) * C_;
  const bf16* ar1 = ar0 + 16 * C_;
  const bf16* wrow[4];
#pragma unroll
  for (int j = 0; j < 4; ++j) wrow[j] = wab + (size_t)(n0 + j * 16 + l16) * C_;

  v8f acc[2][4] = {};
  v16bf a0A, a1A, bA[4], a0B, a1B, bB[4];
  a0A = load_a_frag(ar0, 0, hf);
  a1A = load_a_frag(ar1, 0, hf);
#pragma unroll
  for (int j = 0; j < 4; ++j) bA[j] = load_b_frag(wrow[j], 0, hf);

  for (int k0 = 0; k0 < C_; k0 += 64) {
    const int kB = k0 + 32;
    a0B = load_a_frag(ar0, kB, hf);
    a1B = load_a_frag(ar1, kB, hf);
#pragma unroll
    for (int j = 0; j < 4; ++j) bB[j] = load_b_frag(wrow[j], kB, hf);
#pragma unroll
    for (int j = 0; j < 4; ++j) {
      acc[0][j] = wmma_bf16(a0A, bA[j], acc[0][j]);
      acc[1][j] = wmma_bf16(a1A, bA[j], acc[1][j]);
    }
    const int kA = (k0 + 64 < C_) ? k0 + 64 : 0;  // clamped dummy on last iter
    a0A = load_a_frag(ar0, kA, hf);
    a1A = load_a_frag(ar1, kA, hf);
#pragma unroll
    for (int j = 0; j < 4; ++j) bA[j] = load_b_frag(wrow[j], kA, hf);
#pragma unroll
    for (int j = 0; j < 4; ++j) {
      acc[0][j] = wmma_bf16(a0B, bB[j], acc[0][j]);
      acc[1][j] = wmma_bf16(a1B, bB[j], acc[1][j]);
    }
  }

  // epilogue: C-layout element (r,lane) = (row +r+8*hf, col n0+j*16+l16)
#pragma unroll
  for (int rr = 0; rr < 2; ++rr) {
#pragma unroll
    for (int j = 0; j < 4; ++j) {
#pragma unroll
      for (int r = 0; r < 8; ++r) {
        int m = mt * 32 + rr * 16 + r + 8 * hf;  // token 0..8191
        int n = n0 + j * 16 + l16;               // 0..1535
        bf16 v = (bf16)acc[rr][j][r];
        int bi = m >> 11, t = m & (T_ - 1);
        int sel = n >> 9, inner = n & (C_ - 1);
        int h = inner >> 6, d = inner & (HS - 1);
        size_t bh = (size_t)(bi * H_ + h);
        if (sel == 0)       q[(bh * T_ + t) * HS + d] = v;
        else if (sel == 1)  kk[(bh * T_ + t) * HS + d] = v;
        else                vT[(bh * HS + d) * T_ + t] = v;
      }
    }
  }
}

// ---------------------------------------------------------------------------
// Causal flash attention, transposed-score formulation. One wave per
// (b, h, 16-query tile): S^T = K_tile . Q^T, online softmax over the
// VGPR/half-wave (key) axis, O^T = V^T . P^T.  4096 waves, no LDS/barriers.
// Ping-pong K fragments; next-step K frags + current-step V frags issue right
// after the score WMMAs so their latency hides under the softmax VALU work.
// ---------------------------------------------------------------------------
__global__ __launch_bounds__(256) void attn_kernel(
    const bf16* __restrict__ q, const bf16* __restrict__ kk,
    const bf16* __restrict__ vT, bf16* __restrict__ y) {
  const int lane = threadIdx.x & 31;
  const int wid = (blockIdx.x * blockDim.x + threadIdx.x) >> 5;
  if (wid >= B_ * H_ * (T_ / 16)) return;
  const int qt = wid & (T_ / 16 - 1);
  const int h = (wid >> 7) & (H_ - 1);
  const int b = wid >> 10;
  const int l16 = lane & 15, hl = lane >> 4;
  const size_t bh = (size_t)(b * H_ + h);
  const bf16* qb = q + bh * T_ * HS;
  const bf16* kb = kk + bh * T_ * HS;
  const bf16* vtb = vT + bh * HS * T_;
  const int qglob = qt * 16 + l16;  // this lane's query column

  // Q^T as B-fragments (contraction over hs = 64 -> two fragments)
  v16bf bq0 = load_b_frag(qb + (size_t)qglob * HS, 0, hl);
  v16bf bq1 = load_b_frag(qb + (size_t)qglob * HS, 32, hl);

  v8f ot[4] = {};                 // O^T accum: dim = t*16 + r + 8*hl, col = query
  float M = -__builtin_inff();
  float L = 0.f;

  const int nsteps = (qt * 16 + 47) >> 5;  // cover keys 0 .. qt*16+15

  // one flash step: scores from (c00..c11), prefetch step s+1 into (n00..n11)
  auto step = [&](int s, const v16bf& c00, const v16bf& c01, const v16bf& c10,
                  const v16bf& c11, v16bf& n00, v16bf& n01, v16bf& n10,
                  v16bf& n11) {
    const int j0 = s * 32;
    // S^T tiles: rows = keys, cols = queries
    v8f st0 = {}, st1 = {};
    st0 = wmma_bf16(c00, bq0, st0);
    st0 = wmma_bf16(c01, bq1, st0);
    st1 = wmma_bf16(c10, bq0, st1);
    st1 = wmma_bf16(c11, bq1, st1);

    // issue next-step K loads (clamped, branch-free) + current V loads now;
    // they complete while the softmax VALU below runs
    const int j0n = (s + 1 < nsteps) ? (j0 + 32) : j0;
    n00 = load_a_frag(kb + (size_t)(j0n + l16) * HS, 0, hl);
    n01 = load_a_frag(kb + (size_t)(j0n + l16) * HS, 32, hl);
    n10 = load_a_frag(kb + (size_t)(j0n + 16 + l16) * HS, 0, hl);
    n11 = load_a_frag(kb + (size_t)(j0n + 16 + l16) * HS, 32, hl);
    v16bf av[4];
#pragma unroll
    for (int t = 0; t < 4; ++t)
      av[t] = load_a_frag(vtb + (size_t)(t * 16 + l16) * T_, j0, hl);

    // scale, causal mask, per-column (query) online softmax
    float p0[8], p1[8];
    float mx = -__builtin_inff();
#pragma unroll
    for (int r = 0; r < 8; ++r) {
      int key0 = j0 + hl * 8 + r;   // C-layout row m = r + 8*hl
      float s0 = st0[r] * SM_SCALE;
      float s1 = st1[r] * SM_SCALE;
      if (key0 > qglob) s0 = -__builtin_inff();
      if (key0 + 16 > qglob) s1 = -__builtin_inff();
      p0[r] = s0; p1[r] = s1;
      mx = fmaxf(mx, fmaxf(s0, s1));
    }
    mx = fmaxf(mx, __shfl_xor(mx, 16, 32));  // combine the two key half-groups
    float newM = fmaxf(M, mx);
    float alpha = __expf(M - newM);
    float sum = 0.f;
#pragma unroll
    for (int r = 0; r < 8; ++r) {
      p0[r] = __expf(p0[r] - newM);
      p1[r] = __expf(p1[r] - newM);
      sum += p0[r] + p1[r];
    }
    sum += __shfl_xor(sum, 16, 32);
    L = L * alpha + sum;
    M = newM;

    // Build P^T as a B-fragment (32 keys x 16 queries): only a half-wave swap.
    v16bf pb;
#pragma unroll
    for (int r = 0; r < 8; ++r) {
      float other = hl ? p0[r] : p1[r];          // value the partner lane needs
      float par = __shfl_xor(other, 16, 32);     // partner's value we need
      float elo = hl ? par : p0[r];
      float ehi = hl ? p1[r] : par;
      pb[r] = (bf16)elo;
      pb[r + 8] = (bf16)ehi;
    }

    // O^T += V^T(64 x 32keys) . P^T(32keys x 16)
#pragma unroll
    for (int t = 0; t < 4; ++t) {
#pragma unroll
      for (int r = 0; r < 8; ++r) ot[t][r] *= alpha;
      ot[t] = wmma_bf16(av[t], pb, ot[t]);
    }
  };

  // preload step-0 K fragments (rows = keys) into set A
  v16bf A00 = load_a_frag(kb + (size_t)l16 * HS, 0, hl);
  v16bf A01 = load_a_frag(kb + (size_t)l16 * HS, 32, hl);
  v16bf A10 = load_a_frag(kb + (size_t)(16 + l16) * HS, 0, hl);
  v16bf A11 = load_a_frag(kb + (size_t)(16 + l16) * HS, 32, hl);
  v16bf B00, B01, B10, B11;

  int s = 0;
  while (true) {
    step(s, A00, A01, A10, A11, B00, B01, B10, B11);  // use A, prefetch into B
    if (++s == nsteps) break;
    step(s, B00, B01, B10, B11, A00, A01, A10, A11);  // use B, prefetch into A
    if (++s == nsteps) break;
  }

  const float linv = 1.0f / L;
  const size_t tok = (size_t)(b * T_ + qt * 16 + l16);
#pragma unroll
  for (int t = 0; t < 4; ++t) {
#pragma unroll
    for (int r = 0; r < 8; ++r) {
      int d = t * 16 + r + 8 * hl;
      y[tok * C_ + h * HS + d] = (bf16)(ot[t][r] * linv);
    }
  }
}

// ---------------------------------------------------------------------------
// out = y @ w_proj^T  (fp32 result straight to d_out). 32x64 per wave,
// ping-pong pipeline. 256 * 8 = 2048 waves.
// ---------------------------------------------------------------------------
__global__ __launch_bounds__(256) void proj_gemm_kernel(
    const bf16* __restrict__ yb, const bf16* __restrict__ wpb,
    float* __restrict__ out) {
  const int lane = threadIdx.x & 31;
  const int wid = (blockIdx.x * blockDim.x + threadIdx.x) >> 5;
  const int NT = C_ / 64;  // 8
  const int mt = wid / NT;
  const int n0 = (wid % NT) * 64;
  if (mt >= NTOK / 32) return;
  const int l16 = lane & 15, hf = lane >> 4;
  const bf16* ar0 = yb + (size_t)(mt * 32 + l16) * C_;
  const bf16* ar1 = ar0 + 16 * C_;
  const bf16* wrow[4];
#pragma unroll
  for (int j = 0; j < 4; ++j) wrow[j] = wpb + (size_t)(n0 + j * 16 + l16) * C_;

  v8f acc[2][4] = {};
  v16bf a0A, a1A, bA[4], a0B, a1B, bB[4];
  a0A = load_a_frag(ar0, 0, hf);
  a1A = load_a_frag(ar1, 0, hf);
#pragma unroll
  for (int j = 0; j < 4; ++j) bA[j] = load_b_frag(wrow[j], 0, hf);

  for (int k0 = 0; k0 < C_; k0 += 64) {
    const int kB = k0 + 32;
    a0B = load_a_frag(ar0, kB, hf);
    a1B = load_a_frag(ar1, kB, hf);
#pragma unroll
    for (int j = 0; j < 4; ++j) bB[j] = load_b_frag(wrow[j], kB, hf);
#pragma unroll
    for (int j = 0; j < 4; ++j) {
      acc[0][j] = wmma_bf16(a0A, bA[j], acc[0][j]);
      acc[1][j] = wmma_bf16(a1A, bA[j], acc[1][j]);
    }
    const int kA = (k0 + 64 < C_) ? k0 + 64 : 0;  // clamped dummy on last iter
    a0A = load_a_frag(ar0, kA, hf);
    a1A = load_a_frag(ar1, kA, hf);
#pragma unroll
    for (int j = 0; j < 4; ++j) bA[j] = load_b_frag(wrow[j], kA, hf);
#pragma unroll
    for (int j = 0; j < 4; ++j) {
      acc[0][j] = wmma_bf16(a0B, bB[j], acc[0][j]);
      acc[1][j] = wmma_bf16(a1B, bB[j], acc[1][j]);
    }
  }

#pragma unroll
  for (int rr = 0; rr < 2; ++rr) {
#pragma unroll
    for (int j = 0; j < 4; ++j) {
#pragma unroll
      for (int r = 0; r < 8; ++r) {
        size_t row = (size_t)(mt * 32 + rr * 16 + r + 8 * hf);
        out[row * C_ + n0 + j * 16 + l16] = acc[rr][j][r];
      }
    }
  }
}

extern "C" void kernel_launch(void* const* d_in, const int* in_sizes, int n_in,
                              void* d_out, int out_size, void* d_ws, size_t ws_size,
                              hipStream_t stream) {
  const float* x = (const float*)d_in[0];
  const float* wa = (const float*)d_in[1];
  const float* wp = (const float*)d_in[2];
  float* out = (float*)d_out;

  char* ws = (char*)d_ws;
  size_t off = 0;
  auto alloc = [&](size_t bytes) -> void* {
    void* p = ws + off;
    off += (bytes + 255) & ~(size_t)255;
    return p;
  };
  bf16* xb  = (bf16*)alloc((size_t)NTOK * C_ * 2);       // 8 MB
  bf16* wab = (bf16*)alloc((size_t)3 * C_ * C_ * 2);     // 1.5 MB
  bf16* wpb = (bf16*)alloc((size_t)C_ * C_ * 2);         // 0.5 MB
  bf16* q   = (bf16*)alloc((size_t)NTOK * C_ * 2);       // 8 MB  [B,H,T,hs]
  bf16* kk  = (bf16*)alloc((size_t)NTOK * C_ * 2);       // 8 MB  [B,H,T,hs]
  bf16* vT  = (bf16*)alloc((size_t)NTOK * C_ * 2);       // 8 MB  [B,H,hs,T]
  bf16* yb  = (bf16*)alloc((size_t)NTOK * C_ * 2);       // 8 MB  [B,T,C]

  float* l1out = out + (size_t)NTOK * C_;  // scalar tuple element (= +inf)

  convert_kernel<<<2048, 256, 0, stream>>>(x, wa, wp, xb, wab, wpb, l1out);

  // qkv: 256 m-tiles(32 rows) * 24 n-chunks = 6144 waves -> 768 blocks
  qkv_gemm_kernel<<<768, 256, 0, stream>>>(xb, wab, q, kk, vT);

  // attention: 4*8*128 = 4096 waves -> 512 blocks
  attn_kernel<<<512, 256, 0, stream>>>(q, kk, vT, yb);

  // projection: 256 m-tiles * 8 n-chunks = 2048 waves -> 256 blocks
  proj_gemm_kernel<<<256, 256, 0, stream>>>(yb, wpb, out);
}
